// FlashQwenAttention_39204461478533
// MI455X (gfx1250) — compile-verified, matrix operand order checked by reading, not run on hardware
//
#include <hip/hip_runtime.h>

// ---------------- problem constants (from reference) ----------------
constexpr int B_  = 2;
constexpr int S_  = 1024;
constexpr int H_  = 4096;
constexpr int NH_ = 32;
constexpr int NKV_= 8;
constexpr int HD_ = 128;
constexpr int M_TOK = B_ * S_;          // 2048 rows (tokens)
constexpr float SOFTMAX_SCALE = 0.08838834764831843f; // 1/sqrt(128)

// ---------------- types ----------------
typedef __attribute__((ext_vector_type(16))) __bf16 bf16x16;
typedef __attribute__((ext_vector_type(8)))  float  v8f;

// CDNA5 async global->LDS copy (ASYNCcnt) if the toolchain declares it.
#if __has_builtin(__builtin_amdgcn_global_load_async_to_lds_b128) && \
    __has_builtin(__builtin_amdgcn_s_wait_asynccnt)
#define HAVE_ASYNC_LDS 1
#else
#define HAVE_ASYNC_LDS 0
#endif

#if HAVE_ASYNC_LDS
// Probe-derived signature: (v4i __device__* src, v4i __shared__* dst, Ii, Ii)
typedef int vsi4 __attribute__((vector_size(16)));
typedef __attribute__((address_space(1))) vsi4 gvsi4;
typedef __attribute__((address_space(3))) vsi4 lvsi4;
#define GAS1(p) ((gvsi4*)(p))
#define LAS3(p) ((lvsi4*)(p))
#endif

__device__ __forceinline__ __bf16 f2bf(float f) {
  unsigned u = __builtin_bit_cast(unsigned, f);
  unsigned r = (u + 0x7FFFu + ((u >> 16) & 1u)) >> 16;   // RNE
  unsigned short h = (unsigned short)r;
  return __builtin_bit_cast(__bf16, h);
}

// 16-bit WMMA fragment load, usable for:
//  - A fragment (16M x 32K) from row-major (M x K) storage
//  - B fragment (32K x 16N) from *transposed* (N x K) storage
// Per the ISA layout, each lane's 16 bf16 are two contiguous 16B runs:
//   k = [koff, koff+8) and [16+koff, 16+koff+8), row = lane&15,
//   koff = 8*(lane>=16).  -> two b128 loads per lane.
// Requires (base + row*ld) to be 16B aligned (ld multiple of 8 elements).
__device__ __forceinline__ bf16x16 load_frag_row16(const __bf16* base, int ld) {
  int lane = threadIdx.x & 31;
  int row  = lane & 15;
  int koff = (lane >> 4) << 3;            // 0 or 8
  const __bf16* p = base + (long)row * ld + koff;
  union { bf16x16 v; uint4 q[2]; } f;
  f.q[0] = *(const uint4*)(p);
  f.q[1] = *(const uint4*)(p + 16);
  return f.v;
}

// C/D fragment store: 16x16 f32. VGPR j holds row j (lanes 0-15) / row 8+j
// (lanes 16-31); lane&15 is the column.
__device__ __forceinline__ void store_frag_c(float* C, long ld, v8f acc) {
  int lane = threadIdx.x & 31;
  int col  = lane & 15;
  int mb   = (lane >> 4) << 3;
#pragma unroll
  for (int j = 0; j < 8; ++j) C[(long)(mb + j) * ld + col] = acc[j];
}

// ---------------- fp32 -> bf16 conversion ----------------
__global__ void cvt_f32_bf16_kernel(const float* __restrict__ src,
                                    __bf16* __restrict__ dst, long n) {
  long i = (long)blockIdx.x * blockDim.x + threadIdx.x;
  long stride = (long)gridDim.x * blockDim.x;
  for (; i < n; i += stride) dst[i] = f2bf(src[i]);
}

// ---------------- bf16 GEMM: C(f32, MxN) = A(MxK) * B(KxN) ----------------
// Block tile 128x128, 256 threads = 8 waves (4 along M x 2 along N),
// each wave computes 32x64 = 2x4 WMMA tiles. K-step 32.
// A tile staged directly (async global->LDS when available); B tile staged
// transposed (N-major) so both operand fragment loads are 2 x ds_load_b128.
__global__ __launch_bounds__(256)
void gemm_bf16_kernel(const __bf16* __restrict__ A, const __bf16* __restrict__ Bm,
                      float* __restrict__ C, int M, int N, int K) {
  __shared__ __bf16 As [128][48];   // [m][k], row stride 96B (16B aligned)
  __shared__ __bf16 BsT[128][48];   // [n][k], transposed B tile

  int tid  = threadIdx.x;
  long m0  = (long)blockIdx.y * 128;
  long n0  = (long)blockIdx.x * 128;
  int wave = tid >> 5;
  int wm   = (wave & 3) * 32;       // wave row offset within block tile
  int wn   = (wave >> 2) * 64;      // wave col offset within block tile

  v8f acc[2][4] = {};

  for (int k0 = 0; k0 < K; k0 += 32) {
    // stage A tile 128x32 (512 x 16B chunks)
    for (int c = tid; c < 512; c += 256) {
      int r = c >> 2, cc = (c & 3) * 8;
      const __bf16* src = &A[(m0 + r) * (long)K + k0 + cc];
#if HAVE_ASYNC_LDS
      __builtin_amdgcn_global_load_async_to_lds_b128(GAS1(src),
                                                     LAS3(&As[r][cc]), 0, 0);
#else
      *(uint4*)&As[r][cc] = *(const uint4*)src;
#endif
    }
    // stage B tile 32x128, transposing into BsT[n][k]
    for (int c = tid; c < 512; c += 256) {
      int r = c >> 4, cc = (c & 15) * 8;
      union { uint4 q; __bf16 e[8]; } t;
      t.q = *(const uint4*)&Bm[(long)(k0 + r) * N + n0 + cc];
#pragma unroll
      for (int e2 = 0; e2 < 8; ++e2) BsT[cc + e2][r] = t.e[e2];
    }
    // hint next K tile into cache
    if (k0 + 32 < K) {
      __builtin_prefetch(&A[(m0 + (tid >> 1)) * (long)K + k0 + 32], 0, 1);
      __builtin_prefetch(&Bm[(long)(k0 + 32 + (tid >> 4)) * N + n0], 0, 1);
    }
#if HAVE_ASYNC_LDS
    __builtin_amdgcn_s_wait_asynccnt(0);
#endif
    __syncthreads();

    bf16x16 af0 = load_frag_row16(&As[wm][0],      48);
    bf16x16 af1 = load_frag_row16(&As[wm + 16][0], 48);
#pragma unroll
    for (int j = 0; j < 4; ++j) {
      bf16x16 bfrag = load_frag_row16(&BsT[wn + j * 16][0], 48);
      acc[0][j] = __builtin_amdgcn_wmma_f32_16x16x32_bf16(
          false, af0, false, bfrag, (short)0, acc[0][j], false, false);
      acc[1][j] = __builtin_amdgcn_wmma_f32_16x16x32_bf16(
          false, af1, false, bfrag, (short)0, acc[1][j], false, false);
    }
    __syncthreads();
  }

#pragma unroll
  for (int i = 0; i < 2; ++i)
#pragma unroll
    for (int j = 0; j < 4; ++j)
      store_frag_c(C + (m0 + wm + i * 16) * (long)N + n0 + wn + j * 16,
                   (long)N, acc[i][j]);
}

// ---------------- per-head RMSNorm + RoPE, f32 in -> bf16 out ----------------
// grid: (S, nheads, B), block: 128 (one thread per head-dim element)
__global__ __launch_bounds__(128)
void normrope_kernel(const float* __restrict__ inp, const float* __restrict__ w,
                     __bf16* __restrict__ out, int nheads) {
  int s = blockIdx.x, h = blockIdx.y, b = blockIdx.z;
  int d = threadIdx.x;
  long base = (((long)b * S_ + s) * nheads + h) * HD_;

  float t = inp[base + d];
  float ss = t * t;
#pragma unroll
  for (int m = 16; m; m >>= 1) ss += __shfl_xor(ss, m, 32);
  __shared__ float red[4];
  if ((threadIdx.x & 31) == 0) red[threadIdx.x >> 5] = ss;
  __syncthreads();
  float var = (red[0] + red[1] + red[2] + red[3]) * (1.0f / (float)HD_);
  float rn = t * rsqrtf(var + 1e-6f) * w[d];

  __shared__ float sh[HD_];
  sh[d] = rn;
  __syncthreads();
  float other = sh[d ^ 64];

  int i = d & 63;
  float inv_freq = __powf(10000.0f, -(float)i * (1.0f / 64.0f));
  float ang = (float)s * inv_freq;
  float sn, cs;
  __sincosf(ang, &sn, &cs);
  float r = (d < 64) ? (rn * cs - other * sn) : (other * sn + rn * cs);
  out[base + d] = f2bf(r);
}

// ---------------- flash attention (causal, GQA 32Q/8KV heads) ----------------
// grid: (S/64, NH, B), block 128 = 4 waves; wave w owns q rows [q0+16w, +16).
// K staged [key][dim] (used transposed for QK^T -> b128 frag loads);
// V staged transposed [dim][key] so PV B-fragments are also b128 loads.
__global__ __launch_bounds__(128)
void flash_attn_kernel(const __bf16* __restrict__ Q, const __bf16* __restrict__ Kg,
                       const __bf16* __restrict__ Vg, __bf16* __restrict__ ctx) {
  int q0  = blockIdx.x * 64;
  int h   = blockIdx.y;
  int b   = blockIdx.z;
  int kvh = h >> 2;                      // NH/NKV = 4
  int tid = threadIdx.x;
  int w   = tid >> 5;
  int lane = tid & 31;

  __shared__ __bf16 Ks [32][HD_ + 8];    // [key][dim], row stride 272B
  __shared__ __bf16 VsT[HD_][40];        // [dim][key], row stride 80B
  __shared__ __bf16 Ps[4][16][40];       // per-wave P transpose buffer

  // Q A-fragments (16 rows x 128 dims = 4 frags), kept in registers
  bf16x16 qf[4];
  {
    const __bf16* qbase =
        Q + (((long)b * S_ + (q0 + w * 16)) * NH_ + h) * HD_;
#pragma unroll
    for (int d = 0; d < 4; ++d)
      qf[d] = load_frag_row16(qbase + d * 32, NH_ * HD_);
  }

  v8f o[8] = {};                         // 16 rows x 128 dims accum (C layout)
  float rowM[8], rowL[8];
#pragma unroll
  for (int j = 0; j < 8; ++j) { rowM[j] = -1e30f; rowL[j] = 0.0f; }

  int kcol = lane & 15;
  int qrow_base = q0 + w * 16 + ((lane >> 4) << 3);

  for (int kb = 0; kb < q0 + 64; kb += 32) {
    // stage 32 keys: K directly (async if available), V transposed
    for (int c = tid; c < 512; c += 128) {
      int r = c >> 4, cc = (c & 15) * 8;
      long gidx = (((long)b * S_ + kb + r) * NKV_ + kvh) * HD_ + cc;
#if HAVE_ASYNC_LDS
      __builtin_amdgcn_global_load_async_to_lds_b128(GAS1(&Kg[gidx]),
                                                     LAS3(&Ks[r][cc]), 0, 0);
#else
      *(uint4*)&Ks[r][cc] = *(const uint4*)&Kg[gidx];
#endif
      union { uint4 q; __bf16 e[8]; } t;
      t.q = *(const uint4*)&Vg[gidx];
#pragma unroll
      for (int e2 = 0; e2 < 8; ++e2) VsT[cc + e2][r] = t.e[e2];
    }
#if HAVE_ASYNC_LDS
    __builtin_amdgcn_s_wait_asynccnt(0);
#endif
    __syncthreads();

    // scores: S(16x32) = Q(16x128) * K^T(128x32); B-frags from [key][dim]
    v8f sc0 = {}, sc1 = {};
#pragma unroll
    for (int d = 0; d < 4; ++d) {
      bf16x16 kf0 = load_frag_row16(&Ks[0][d * 32],  HD_ + 8);
      bf16x16 kf1 = load_frag_row16(&Ks[16][d * 32], HD_ + 8);
      sc0 = __builtin_amdgcn_wmma_f32_16x16x32_bf16(
          false, qf[d], false, kf0, (short)0, sc0, false, false);
      sc1 = __builtin_amdgcn_wmma_f32_16x16x32_bf16(
          false, qf[d], false, kf1, (short)0, sc1, false, false);
    }

    // online softmax per row (row = qrow_base + j; 16 lanes of a half share it)
#pragma unroll
    for (int j = 0; j < 8; ++j) {
      int qidx = qrow_base + j;
      float s0 = sc0[j] * SOFTMAX_SCALE;
      float s1 = sc1[j] * SOFTMAX_SCALE;
      if (kb + kcol > qidx)      s0 = -1e30f;   // causal mask
      if (kb + 16 + kcol > qidx) s1 = -1e30f;
      float t = fmaxf(s0, s1);
#pragma unroll
      for (int m = 1; m < 16; m <<= 1) t = fmaxf(t, __shfl_xor(t, m, 32));
      float nm   = fmaxf(rowM[j], t);
      float corr = __expf(rowM[j] - nm);
      float p0 = __expf(s0 - nm);
      float p1 = __expf(s1 - nm);
      float rs = p0 + p1;
#pragma unroll
      for (int m = 1; m < 16; m <<= 1) rs += __shfl_xor(rs, m, 32);
      rowL[j] = rowL[j] * corr + rs;
      rowM[j] = nm;
#pragma unroll
      for (int d = 0; d < 8; ++d) o[d][j] *= corr;
      int prow = ((lane >> 4) << 3) + j;    // C-layout row -> A-layout row
      Ps[w][prow][kcol]      = f2bf(p0);
      Ps[w][prow][kcol + 16] = f2bf(p1);
    }
    __syncthreads();

    // O += P(16x32) * V(32x128); V B-frags from transposed [dim][key]
    bf16x16 pf = load_frag_row16(&Ps[w][0][0], 40);
#pragma unroll
    for (int d = 0; d < 8; ++d) {
      bf16x16 vfrag = load_frag_row16(&VsT[d * 16][0], 40);
      o[d] = __builtin_amdgcn_wmma_f32_16x16x32_bf16(
          false, pf, false, vfrag, (short)0, o[d], false, false);
    }
    __syncthreads();
  }

  // epilogue: normalize by row sum, write ctx as bf16 (token-major, NH*HD)
#pragma unroll
  for (int j = 0; j < 8; ++j) {
    float inv = 1.0f / rowL[j];
    long obase = ((long)b * S_ + (qrow_base + j)) * (long)(NH_ * HD_)
               + (long)h * HD_;
#pragma unroll
    for (int d = 0; d < 8; ++d)
      ctx[obase + d * 16 + kcol] = f2bf(o[d][j] * inv);
  }
}

// ---------------- host side ----------------
extern "C" void kernel_launch(void* const* d_in, const int* in_sizes, int n_in,
                              void* d_out, int out_size, void* d_ws, size_t ws_size,
                              hipStream_t stream) {
  const float* x   = (const float*)d_in[0];
  const float* wq  = (const float*)d_in[1];
  const float* wk  = (const float*)d_in[2];
  const float* wv  = (const float*)d_in[3];
  const float* wo  = (const float*)d_in[4];
  const float* qnw = (const float*)d_in[5];
  const float* knw = (const float*)d_in[6];
  float* out = (float*)d_out;

  char* ws = (char*)d_ws;
  auto alloc = [&](size_t bytes) {
    char* p = ws;
    ws += (bytes + 255) & ~(size_t)255;
    return p;
  };

  const size_t N_X   = (size_t)M_TOK * H_;            // 8.4M
  const size_t N_WQ  = (size_t)H_ * (NH_ * HD_);      // 16.8M
  const size_t N_WKV = (size_t)H_ * (NKV_ * HD_);     // 4.2M
  const size_t N_Q   = (size_t)M_TOK * (NH_ * HD_);   // 8.4M
  const size_t N_KV  = (size_t)M_TOK * (NKV_ * HD_);  // 2.1M

  __bf16* xb   = (__bf16*)alloc(N_X  * 2);
  __bf16* wqb  = (__bf16*)alloc(N_WQ * 2);
  __bf16* wkb  = (__bf16*)alloc(N_WKV * 2);
  __bf16* wvb  = (__bf16*)alloc(N_WKV * 2);
  __bf16* wob  = (__bf16*)alloc(N_WQ * 2);
  float*  qf   = (float*) alloc(N_Q  * 4);
  float*  kf   = (float*) alloc(N_KV * 4);
  float*  vf   = (float*) alloc(N_KV * 4);
  __bf16* qbn  = (__bf16*)alloc(N_Q  * 2);
  __bf16* kbn  = (__bf16*)alloc(N_KV * 2);
  __bf16* vbn  = (__bf16*)alloc(N_KV * 2);
  __bf16* ctxb = (__bf16*)alloc(N_Q  * 2);

  // 1) fp32 -> bf16 conversions
  cvt_f32_bf16_kernel<<<4096, 256, 0, stream>>>(x,  xb,  (long)N_X);
  cvt_f32_bf16_kernel<<<4096, 256, 0, stream>>>(wq, wqb, (long)N_WQ);
  cvt_f32_bf16_kernel<<<4096, 256, 0, stream>>>(wk, wkb, (long)N_WKV);
  cvt_f32_bf16_kernel<<<4096, 256, 0, stream>>>(wv, wvb, (long)N_WKV);
  cvt_f32_bf16_kernel<<<4096, 256, 0, stream>>>(wo, wob, (long)N_WQ);

  // 2) QKV projections (all dims multiples of 128)
  gemm_bf16_kernel<<<dim3(32, 16), 256, 0, stream>>>(xb, wqb, qf,
      M_TOK, NH_ * HD_, H_);
  gemm_bf16_kernel<<<dim3(8, 16), 256, 0, stream>>>(xb, wkb, kf,
      M_TOK, NKV_ * HD_, H_);
  gemm_bf16_kernel<<<dim3(8, 16), 256, 0, stream>>>(xb, wvb, vf,
      M_TOK, NKV_ * HD_, H_);

  // 3) per-head RMSNorm + RoPE for Q and K; V just converts to bf16
  normrope_kernel<<<dim3(S_, NH_, B_), 128, 0, stream>>>(qf, qnw, qbn, NH_);
  normrope_kernel<<<dim3(S_, NKV_, B_), 128, 0, stream>>>(kf, knw, kbn, NKV_);
  cvt_f32_bf16_kernel<<<2048, 256, 0, stream>>>(vf, vbn, (long)N_KV);

  // 4) causal GQA flash attention -> ctx (bf16)
  flash_attn_kernel<<<dim3(S_ / 64, NH_, B_), 128, 0, stream>>>(qbn, kbn, vbn, ctxb);

  // 5) output projection -> fp32 result
  gemm_bf16_kernel<<<dim3(32, 16), 256, 0, stream>>>(ctxb, wob, out,
      M_TOK, H_, NH_ * HD_);

  (void)in_sizes; (void)n_in; (void)out_size; (void)ws_size;
}
